// MultiHeadAttention_59691455480063
// MI455X (gfx1250) — matrix operand order repaired and checked by reading, not compiled, not run on hardware
//
#include <hip/hip_runtime.h>
#include <hip/hip_bf16.h>

// ---------------------------------------------------------------------------
// MI455X (gfx1250) multi-head attention forward.
// All matmuls via v_wmma_f32_16x16x32_bf16 (wave32 WMMA, f32 accumulate).
// ---------------------------------------------------------------------------

typedef __bf16 bf16_t;
typedef __bf16 v16bf __attribute__((ext_vector_type(16)));
typedef float  v8f   __attribute__((ext_vector_type(8)));

union FragB16 {
    uint4  u[2];   // 32 bytes = 16 bf16
    v16bf  v;
    bf16_t h[16];
};

constexpr int Bsz = 2;
constexpr int Sq  = 2048;
constexpr int Dm  = 1024;
constexpr int Hh  = 16;
constexpr int Dk  = 64;

// ---------------------------------------------------------------------------
// GEMM: Y = X[M,K](f32) @ W[N,K](f32)^T + bias   (torch Linear convention)
// Block tile 128x64, 8 waves, each wave 32x32 (2x2 WMMA frags), K-step 32.
// mode 0: write bf16 scattered to [B,H,S,dk]   (QKV projections)
// mode 1: write f32 row-major [M,N]            (output projection)
// ---------------------------------------------------------------------------
#define GBM 128
#define GBN 64
#define GBK 32

__global__ __launch_bounds__(256) void gemm_xwt_bias(
    const float* __restrict__ X, const float* __restrict__ W,
    const float* __restrict__ bias, void* __restrict__ Y,
    int M, int N, int K, int mode)
{
    __shared__ __align__(16) bf16_t As[GBM][GBK];   // m-major, 8 KB
    __shared__ __align__(16) bf16_t Bs[GBK][GBN];   // k-major (W transposed), 4 KB

    const int tid  = threadIdx.x;
    const int lane = tid & 31;
    const int wave = tid >> 5;      // 0..7
    const int wm   = wave >> 1;     // 0..3 -> 32-row strip
    const int wn   = wave & 1;      // 0..1 -> 32-col strip

    const int mBase = blockIdx.y * GBM;
    const int nBase = blockIdx.x * GBN;

    v8f c[2][2];
#pragma unroll
    for (int i = 0; i < 2; ++i)
#pragma unroll
        for (int j = 0; j < 2; ++j)
#pragma unroll
            for (int r = 0; r < 8; ++r) c[i][j][r] = 0.0f;

    const int arow  = tid >> 1;          // 0..127
    const int ahalf = (tid & 1) * 16;    // 0 / 16
    const int brow  = tid >> 2;          // 0..63
    const int bcg   = (tid & 3) * 8;     // 0,8,16,24
    const int koff  = (lane < 16) ? 0 : 8;

    for (int k0 = 0; k0 < K; k0 += GBK) {
        // --- stage A tile: 16 f32 per thread, convert to bf16 ---
        const float* ap = X + (size_t)(mBase + arow) * K + k0 + ahalf;
        float4 a0 = ((const float4*)ap)[0];
        float4 a1 = ((const float4*)ap)[1];
        float4 a2 = ((const float4*)ap)[2];
        float4 a3 = ((const float4*)ap)[3];
        FragB16 fa;
        fa.h[0]=(bf16_t)a0.x; fa.h[1]=(bf16_t)a0.y; fa.h[2]=(bf16_t)a0.z; fa.h[3]=(bf16_t)a0.w;
        fa.h[4]=(bf16_t)a1.x; fa.h[5]=(bf16_t)a1.y; fa.h[6]=(bf16_t)a1.z; fa.h[7]=(bf16_t)a1.w;
        fa.h[8]=(bf16_t)a2.x; fa.h[9]=(bf16_t)a2.y; fa.h[10]=(bf16_t)a2.z; fa.h[11]=(bf16_t)a2.w;
        fa.h[12]=(bf16_t)a3.x; fa.h[13]=(bf16_t)a3.y; fa.h[14]=(bf16_t)a3.z; fa.h[15]=(bf16_t)a3.w;
        ((uint4*)&As[arow][ahalf])[0] = fa.u[0];
        ((uint4*)&As[arow][ahalf])[1] = fa.u[1];

        // --- stage B tile: 8 f32 per thread, transposed store (k-major) ---
        const float* wp = W + (size_t)(nBase + brow) * K + k0 + bcg;
        float4 w0 = ((const float4*)wp)[0];
        float4 w1 = ((const float4*)wp)[1];
        const float wv[8] = {w0.x, w0.y, w0.z, w0.w, w1.x, w1.y, w1.z, w1.w};
#pragma unroll
        for (int j = 0; j < 8; ++j) Bs[bcg + j][brow] = (bf16_t)wv[j];

        if (k0 + GBK < K) {                         // global_prefetch_b8 next tile
            __builtin_prefetch(ap + GBK, 0, 1);
            __builtin_prefetch(wp + GBK, 0, 1);
        }
        __syncthreads();

        // --- fragments + 4 WMMAs ---
        FragB16 a[2], b[2];
#pragma unroll
        for (int mi = 0; mi < 2; ++mi) {
            const int ar = wm * 32 + mi * 16 + (lane & 15);
            a[mi].u[0] = *(const uint4*)&As[ar][koff];
            a[mi].u[1] = *(const uint4*)&As[ar][koff + 16];
        }
#pragma unroll
        for (int ni = 0; ni < 2; ++ni) {
            const int bc = wn * 32 + ni * 16;
#pragma unroll
            for (int s = 0; s < 16; ++s)
                b[ni].h[s] = Bs[koff + (s & 7) + (s >> 3) * 16][bc + (lane & 15)];
        }
#pragma unroll
        for (int mi = 0; mi < 2; ++mi)
#pragma unroll
            for (int ni = 0; ni < 2; ++ni)
                c[mi][ni] = __builtin_amdgcn_wmma_f32_16x16x32_bf16(
                    false, a[mi].v, false, b[ni].v, (short)0, c[mi][ni],
                    false, false);
        __syncthreads();
    }

    // --- epilogue: bias + store ---
    const int rowadd = (lane < 16) ? 0 : 8;
#pragma unroll
    for (int mi = 0; mi < 2; ++mi) {
#pragma unroll
        for (int ni = 0; ni < 2; ++ni) {
            const int nglob = nBase + wn * 32 + ni * 16 + (lane & 15);
            const float bv = bias[nglob];
#pragma unroll
            for (int r = 0; r < 8; ++r) {
                const int mglob = mBase + wm * 32 + mi * 16 + r + rowadd;
                const float val = c[mi][ni][r] + bv;
                if (mode == 0) {
                    // scatter to [B,H,S,dk] bf16
                    const int bt = mglob >> 11, s = mglob & (Sq - 1);
                    const int h  = nglob >> 6,  d = nglob & (Dk - 1);
                    ((bf16_t*)Y)[(((size_t)(bt * Hh + h) * Sq) + s) * Dk + d] =
                        (bf16_t)val;
                } else {
                    ((float*)Y)[(size_t)mglob * N + nglob] = val;
                }
            }
        }
    }
}

// ---------------------------------------------------------------------------
// Flash attention: one block per (b*h, 64-query tile); 4 waves x 16 q-rows.
// Per 32-key step: QK^T (4 WMMA) -> online softmax -> P@V (4 WMMA).
// No 1/sqrt(dk) scale (faithful to reference). O written f32 [B*S, D].
// ---------------------------------------------------------------------------
__global__ __launch_bounds__(128) void flash_attn_bf16(
    const bf16_t* __restrict__ Qp, const bf16_t* __restrict__ Kp,
    const bf16_t* __restrict__ Vp, float* __restrict__ O)
{
    __shared__ __align__(16) bf16_t Kt[Dk][32];      // K tile transposed, 4 KB
    __shared__ __align__(16) bf16_t Vs[32][Dk];      // V tile as-is,      4 KB
    __shared__ __align__(16) bf16_t Ps[4][16][32];   // per-wave P tiles,  4 KB

    const int tid  = threadIdx.x;
    const int lane = tid & 31;
    const int wave = tid >> 5;       // 0..3
    const int bh    = blockIdx.y;
    const int batch = bh >> 4;
    const int head  = bh & 15;

    const bf16_t* Qh = Qp + (size_t)bh * Sq * Dk;
    const bf16_t* Kh = Kp + (size_t)bh * Sq * Dk;
    const bf16_t* Vh = Vp + (size_t)bh * Sq * Dk;

    const int qWave = blockIdx.x * 64 + wave * 16;
    const int koff  = (lane < 16) ? 0 : 8;

    // Q fragments for this wave's 16 rows (dk=64 -> 2 A-frags)
    FragB16 qf[2];
    {
        const bf16_t* qp0 = Qh + (size_t)(qWave + (lane & 15)) * Dk;
#pragma unroll
        for (int t = 0; t < 2; ++t) {
            qf[t].u[0] = *(const uint4*)(qp0 + t * 32 + koff);
            qf[t].u[1] = *(const uint4*)(qp0 + t * 32 + koff + 16);
        }
    }

    v8f o[4];
#pragma unroll
    for (int j = 0; j < 4; ++j)
#pragma unroll
        for (int r = 0; r < 8; ++r) o[j][r] = 0.0f;
    float mrun[8], lrun[8];
#pragma unroll
    for (int r = 0; r < 8; ++r) { mrun[r] = -3.0e38f; lrun[r] = 0.0f; }

    const int ldr = tid >> 2;          // 0..31  key row
    const int ldd = (tid & 3) * 16;    // 0,16,32,48

    for (int kb = 0; kb < Sq; kb += 32) {
        // ---- stage K (transposed) and V tiles ----
        {
            const bf16_t* kp0 = Kh + (size_t)(kb + ldr) * Dk + ldd;
            FragB16 kv;
            kv.u[0] = ((const uint4*)kp0)[0];
            kv.u[1] = ((const uint4*)kp0)[1];
#pragma unroll
            for (int j = 0; j < 16; ++j) Kt[ldd + j][ldr] = kv.h[j];
            const bf16_t* vp0 = Vh + (size_t)(kb + ldr) * Dk + ldd;
            ((uint4*)&Vs[ldr][ldd])[0] = ((const uint4*)vp0)[0];
            ((uint4*)&Vs[ldr][ldd])[1] = ((const uint4*)vp0)[1];
        }
        __syncthreads();

        // ---- scores: S[16q x 32k] = Q @ K^T ----
        v8f sc[2];
#pragma unroll
        for (int ni = 0; ni < 2; ++ni) {
#pragma unroll
            for (int r = 0; r < 8; ++r) sc[ni][r] = 0.0f;
#pragma unroll
            for (int t = 0; t < 2; ++t) {
                FragB16 kf;
#pragma unroll
                for (int s = 0; s < 16; ++s)
                    kf.h[s] = Kt[t * 32 + koff + (s & 7) + (s >> 3) * 16][ni * 16 + (lane & 15)];
                sc[ni] = __builtin_amdgcn_wmma_f32_16x16x32_bf16(
                    false, qf[t].v, false, kf.v, (short)0, sc[ni],
                    false, false);
            }
        }

        // ---- online softmax (row reductions across 16-lane groups) ----
        float scale[8];
#pragma unroll
        for (int r = 0; r < 8; ++r) {
            const float s0 = sc[0][r], s1 = sc[1][r];
            float mx = fmaxf(s0, s1);
#pragma unroll
            for (int off = 1; off < 16; off <<= 1)
                mx = fmaxf(mx, __shfl_xor(mx, off, 16));
            const float mnew = fmaxf(mrun[r], mx);
            scale[r] = __expf(mrun[r] - mnew);
            const float p0 = __expf(s0 - mnew);
            const float p1 = __expf(s1 - mnew);
            float rs = p0 + p1;
#pragma unroll
            for (int off = 1; off < 16; off <<= 1)
                rs += __shfl_xor(rs, off, 16);
            lrun[r] = lrun[r] * scale[r] + rs;
            mrun[r] = mnew;
            const int prow = r + ((lane < 16) ? 0 : 8);
            Ps[wave][prow][lane & 15]        = (bf16_t)p0;
            Ps[wave][prow][16 + (lane & 15)] = (bf16_t)p1;
        }
#pragma unroll
        for (int j = 0; j < 4; ++j)
#pragma unroll
            for (int r = 0; r < 8; ++r) o[j][r] *= scale[r];

        __syncthreads();   // Ps (C-layout -> A-layout via LDS) + tiles settled

        // ---- O += P @ V ----
        FragB16 pa;
        pa.u[0] = *(const uint4*)&Ps[wave][lane & 15][koff];
        pa.u[1] = *(const uint4*)&Ps[wave][lane & 15][koff + 16];
#pragma unroll
        for (int j = 0; j < 4; ++j) {
            FragB16 vf;   // B-frag: column d = lane & 15, elements over key (table order)
#pragma unroll
            for (int s = 0; s < 16; ++s)
                vf.h[s] = Vs[koff + (s & 7) + (s >> 3) * 16][j * 16 + (lane & 15)];
            o[j] = __builtin_amdgcn_wmma_f32_16x16x32_bf16(
                false, pa.v, false, vf.v, (short)0, o[j], false, false);
        }
        __syncthreads();   // done reading Kt/Vs before next stage
    }

    // ---- normalize and write O as f32 [B*S, D] (heads re-interleaved) ----
    const int rowadd = (lane < 16) ? 0 : 8;
#pragma unroll
    for (int r = 0; r < 8; ++r) {
        const float inv = 1.0f / lrun[r];
        const int sq = qWave + r + rowadd;
        const size_t mrow = (size_t)(batch * Sq + sq) * Dm;
#pragma unroll
        for (int j = 0; j < 4; ++j) {
            const int col = head * Dk + j * 16 + (lane & 15);
            O[mrow + col] = o[j][r] * inv;
        }
    }
}

// ---------------------------------------------------------------------------
// Host-side launch
// ---------------------------------------------------------------------------
extern "C" void kernel_launch(void* const* d_in, const int* in_sizes, int n_in,
                              void* d_out, int out_size, void* d_ws, size_t ws_size,
                              hipStream_t stream)
{
    (void)in_sizes; (void)n_in; (void)out_size; (void)ws_size;

    const float* q  = (const float*)d_in[0];
    const float* k  = (const float*)d_in[1];
    const float* v  = (const float*)d_in[2];
    const float* Wq = (const float*)d_in[3];
    const float* bq = (const float*)d_in[4];
    const float* Wk = (const float*)d_in[5];
    const float* bk = (const float*)d_in[6];
    const float* Wv = (const float*)d_in[7];
    const float* bv = (const float*)d_in[8];
    const float* Wo = (const float*)d_in[9];
    const float* bo = (const float*)d_in[10];
    float* out = (float*)d_out;

    const int M = Bsz * Sq;   // 4096
    const int N = Dm;         // 1024
    const int K = Dm;         // 1024

    // workspace carve: Qp/Kp/Vp bf16 [B,H,S,dk] (8 MB each) + O f32 (16 MB)
    char* ws = (char*)d_ws;
    const size_t projBytes = (size_t)M * Dm * sizeof(bf16_t);  // 8 MB
    bf16_t* Qp   = (bf16_t*)(ws);
    bf16_t* Kp   = (bf16_t*)(ws + projBytes);
    bf16_t* Vp   = (bf16_t*)(ws + 2 * projBytes);
    float*  Obuf = (float*)(ws + 3 * projBytes);               // 16 MB

    dim3 gblk(256);
    dim3 ggrd(N / GBN, M / GBM);   // (16, 32)

    gemm_xwt_bias<<<ggrd, gblk, 0, stream>>>(q, Wq, bq, Qp, M, N, K, 0);
    gemm_xwt_bias<<<ggrd, gblk, 0, stream>>>(k, Wk, bk, Kp, M, N, K, 0);
    gemm_xwt_bias<<<ggrd, gblk, 0, stream>>>(v, Wv, bv, Vp, M, N, K, 0);

    dim3 ablk(128);
    dim3 agrd(Sq / 64, Bsz * Hh);  // (32, 32)
    flash_attn_bf16<<<agrd, ablk, 0, stream>>>(Qp, Kp, Vp, Obuf);

    gemm_xwt_bias<<<ggrd, gblk, 0, stream>>>(Obuf, Wo, bo, out, M, N, K, 1);
}
